// DecoderLayer_89567247990907
// MI455X (gfx1250) — matrix operand order repaired and checked by reading, not compiled, not run on hardware
//
#include <hip/hip_runtime.h>
#include <hip/hip_bf16.h>
#include <stdint.h>

// ---------------------------------------------------------------------------
// Types for CDNA5 WMMA (wave32, v_wmma_f32_16x16x32_bf16)
// ---------------------------------------------------------------------------
typedef __bf16 bf16_t;
typedef __attribute__((ext_vector_type(16))) __bf16 v16bf;
typedef __attribute__((ext_vector_type(8)))  __bf16 v8bf;
typedef __attribute__((ext_vector_type(8)))  float  v8f;

#define NEG_SCORE (-1.0e9f)

__device__ __forceinline__ v8f wmma_bf16(v16bf a, v16bf b, v8f c) {
  // (neg_a, A, neg_b, B, c_mod, C, reuse_a, reuse_b)
  return __builtin_amdgcn_wmma_f32_16x16x32_bf16(false, a, false, b, (short)0, c,
                                                 false, false);
}

// Build a 16-element bf16 fragment from two 8-element (16-byte) chunks.
__device__ __forceinline__ v16bf ld_pair8(const bf16_t* p0, const bf16_t* p1) {
  v8bf a = *(const v8bf*)p0;
  v8bf b = *(const v8bf*)p1;
  return __builtin_shufflevector(a, b, 0,1,2,3,4,5,6,7,8,9,10,11,12,13,14,15);
}
__device__ __forceinline__ v16bf ld_contig16(const bf16_t* p) {
  return ld_pair8(p, p + 8);
}

// gfx1250 async global->LDS copy (16 bytes per lane), tracked by ASYNCcnt.
// GVS mode: mem = SADDR(64b) + VADDR(32b offset); LDS addr from VDST VGPR.
__device__ __forceinline__ void async_copy_b128(uint32_t lds_byte_addr,
                                                uint32_t gbl_byte_off,
                                                uint64_t gbl_base) {
  asm volatile("global_load_async_to_lds_b128 %0, %1, %2"
               :
               : "v"(lds_byte_addr), "v"(gbl_byte_off), "s"(gbl_base)
               : "memory");
}
__device__ __forceinline__ void wait_async0() {
  asm volatile("s_wait_asynccnt 0x0" ::: "memory");
}

// ---------------------------------------------------------------------------
// Weight pre-pass: W fp32 [K,N] -> Wt bf16 [N,K]  (64x64 tiles via LDS)
// ---------------------------------------------------------------------------
__global__ __launch_bounds__(256)
void wt_bf16_kernel(const float* __restrict__ W, bf16_t* __restrict__ Wt,
                    int K, int N) {
  __shared__ float tile[64][65];
  const int n0 = blockIdx.x * 64;
  const int k0 = blockIdx.y * 64;
  const int tx = threadIdx.x & 63;
  const int tg = (threadIdx.x >> 6) * 16;  // 4 groups of 16 rows
#pragma unroll
  for (int i = 0; i < 16; i++) {
    const int k = tg + i;
    tile[k][tx] = W[(size_t)(k0 + k) * N + (n0 + tx)];
  }
  __syncthreads();
#pragma unroll
  for (int i = 0; i < 16; i++) {
    const int n = tg + i;
    Wt[(size_t)(n0 + n) * K + (k0 + tx)] = (bf16_t)tile[tx][n];
  }
}

// ---------------------------------------------------------------------------
// LayerNorm -> bf16 (one row per block, D = 1024)
// ---------------------------------------------------------------------------
__global__ __launch_bounds__(256)
void ln_bf16_kernel(const float* __restrict__ x, const float* __restrict__ g,
                    const float* __restrict__ bet, bf16_t* __restrict__ out) {
  const int D_ = 1024;
  const int row = blockIdx.x;
  const float* xr = x + (size_t)row * D_;
  float s = 0.f, s2 = 0.f;
#pragma unroll
  for (int it = 0; it < 4; it++) {
    float v = xr[threadIdx.x + it * 256];
    s += v; s2 += v * v;
  }
#pragma unroll
  for (int off = 16; off >= 1; off >>= 1) {
    s  += __shfl_xor(s,  off, 32);
    s2 += __shfl_xor(s2, off, 32);
  }
  __shared__ float rs[8], rs2[8];
  const int wid = threadIdx.x >> 5, lane = threadIdx.x & 31;
  if (lane == 0) { rs[wid] = s; rs2[wid] = s2; }
  __syncthreads();
  float ts = 0.f, ts2 = 0.f;
#pragma unroll
  for (int i = 0; i < 8; i++) { ts += rs[i]; ts2 += rs2[i]; }
  const float mu   = ts * (1.0f / 1024.0f);
  const float var  = ts2 * (1.0f / 1024.0f) - mu * mu;
  const float rstd = rsqrtf(var + 1e-5f);
  bf16_t* orow = out + (size_t)row * D_;
#pragma unroll
  for (int it = 0; it < 4; it++) {
    const int i = threadIdx.x + it * 256;
    const float v = (xr[i] - mu) * rstd * g[i] + bet[i];
    orow[i] = (bf16_t)v;
  }
}

// ---------------------------------------------------------------------------
// WMMA GEMM with async double-buffered LDS:
//   out = epilogue(A_bf16[M,K] @ Wt_bf16[N,K]^T + bias)
//   optional exact GELU, optional residual add (fp32), fp32 and/or bf16 out.
// Block: 256 threads = 8 waves; tile 128x128x32; wave tile 32x64.
// Both A and B tiles are contiguous bf16 row copies -> global_load_async_to_lds.
// ---------------------------------------------------------------------------
#define GBM 128
#define GBN 128
#define GBK 32
#define LDSA 40  // 32 + 8 pad; 80B row stride keeps 16B alignment + bank spread
#define TILE_BYTES (GBM * LDSA * 2)

__global__ __launch_bounds__(256)
void gemm_bf16_wmma(const bf16_t* __restrict__ A, const bf16_t* __restrict__ Wt,
                    const float* __restrict__ bias, const float* __restrict__ R,
                    float* __restrict__ outF, bf16_t* __restrict__ outB,
                    int M, int N, int K, int gelu) {
  __shared__ __attribute__((aligned(16))) bf16_t ldsA[2][GBM * LDSA];
  __shared__ __attribute__((aligned(16))) bf16_t ldsB[2][GBN * LDSA];

  const int tid  = threadIdx.x;
  const int wid  = tid >> 5;
  const int lane = tid & 31;
  const int l16  = lane & 15;
  const int lh8  = (lane & 16) ? 8 : 0;
  const int lh16 = (lane & 16) ? 16 : 0;

  const int m0 = blockIdx.y * GBM;
  const int n0 = blockIdx.x * GBN;
  const int wm = (wid & 3) * 32;   // 4 waves along M
  const int wn = (wid >> 2) * 64;  // 2 waves along N

  const v8f vzero = {0.f, 0.f, 0.f, 0.f, 0.f, 0.f, 0.f, 0.f};
  v8f acc[2][4];
#pragma unroll
  for (int i = 0; i < 2; i++)
#pragma unroll
    for (int j = 0; j < 4; j++) acc[i][j] = vzero;

  // Async copy mapping: 128 rows x 32 cols bf16 per tile; each thread moves
  // one 32-byte row segment (two b128 transfers) of A and of B.
  const int crow = tid >> 1;           // 0..127
  const int cseg = (tid & 1) * 16;     // bf16 col offset 0 / 16
  const uint32_t aLds0 =
      (uint32_t)(uintptr_t)&ldsA[0][0] + (uint32_t)(crow * LDSA + cseg) * 2u;
  const uint32_t bLds0 =
      (uint32_t)(uintptr_t)&ldsB[0][0] + (uint32_t)(crow * LDSA + cseg) * 2u;
  const uint64_t abase = (uint64_t)(uintptr_t)A;
  const uint64_t bbase = (uint64_t)(uintptr_t)Wt;

  auto prefetch = [&](int k0, int buf) {
    const uint32_t ga =
        (uint32_t)(((size_t)(m0 + crow) * K + (k0 + cseg)) * 2u);
    const uint32_t gb =
        (uint32_t)(((size_t)(n0 + crow) * K + (k0 + cseg)) * 2u);
    const uint32_t bofs = (uint32_t)buf * (uint32_t)TILE_BYTES;
    async_copy_b128(aLds0 + bofs,      ga,      abase);
    async_copy_b128(aLds0 + bofs + 16, ga + 16, abase);
    async_copy_b128(bLds0 + bofs,      gb,      bbase);
    async_copy_b128(bLds0 + bofs + 16, gb + 16, bbase);
  };

  prefetch(0, 0);
  const int T = K / GBK;
  for (int it = 0; it < T; it++) {
    wait_async0();       // my previous async fills have landed in LDS
    __syncthreads();     // everyone's fills landed; prior reads of other buf done
    if (it + 1 < T) prefetch((it + 1) * GBK, (it + 1) & 1);

    const bf16_t* ca = ldsA[it & 1];
    const bf16_t* cb = ldsB[it & 1];

    v16bf af[2];
#pragma unroll
    for (int i = 0; i < 2; i++) {
      const bf16_t* base = &ca[(wm + i * 16 + l16) * LDSA];
      af[i] = ld_pair8(base + lh8, base + 16 + lh8);  // A 16x32 layout
    }
    v16bf bfr[4];
#pragma unroll
    for (int j = 0; j < 4; j++) {
      const bf16_t* base = &cb[(wn + j * 16 + l16) * LDSA];
      bfr[j] = ld_pair8(base + lh8, base + 16 + lh8);  // B 32x16 layout
    }
#pragma unroll
    for (int i = 0; i < 2; i++)
#pragma unroll
      for (int j = 0; j < 4; j++)
        acc[i][j] = wmma_bf16(af[i], bfr[j], acc[i][j]);
  }

  // Epilogue: C layout row = r + lh8, col = l16 within each 16x16 tile.
#pragma unroll
  for (int j = 0; j < 4; j++) {
    const int col = n0 + wn + j * 16 + l16;
    const float bb = bias[col];
#pragma unroll
    for (int i = 0; i < 2; i++) {
#pragma unroll
      for (int r = 0; r < 8; r++) {
        const int row = m0 + wm + i * 16 + r + lh8;
        float v = acc[i][j][r] + bb;
        if (gelu) v = 0.5f * v * (1.0f + erff(v * 0.70710678118654752f));
        if (R)    v += R[(size_t)row * N + col];
        if (outF) outF[(size_t)row * N + col] = v;
        if (outB) outB[(size_t)row * N + col] = (bf16_t)v;
      }
    }
  }
}

// ---------------------------------------------------------------------------
// Flash attention with strided/local mask, WMMA QK^T and PV.
// Block: 128 threads = 4 waves; 64 query rows per block; key tiles of 64.
// Tile skip: key tile at distance d = q0-kt0 contributes iff d<=128 or d%128==0.
// ---------------------------------------------------------------------------
#define LDK 72  // 64 + 8 pad

__global__ __launch_bounds__(128)
void attn_wmma(const bf16_t* __restrict__ Q, const bf16_t* __restrict__ Kg,
               const bf16_t* __restrict__ Vg, const int* __restrict__ umask,
               bf16_t* __restrict__ O, int S, int D, int H) {
  __shared__ __attribute__((aligned(16))) bf16_t ldsK [64 * LDK];      // [key][dk]
  __shared__ __attribute__((aligned(16))) bf16_t ldsVt[64 * LDK];      // [dk][key]
  __shared__ __attribute__((aligned(16))) bf16_t ldsP [4][16 * LDK];   // per-wave P

  const int tid  = threadIdx.x;
  const int wid  = tid >> 5;
  const int lane = tid & 31;
  const int l16  = lane & 15;
  const int lh8  = (lane & 16) ? 8 : 0;
  const int lh16 = (lane & 16) ? 16 : 0;

  const int q0 = blockIdx.x * 64;
  const int bh = blockIdx.y;
  const int b = bh / H, h = bh % H;
  const size_t rowbase = (size_t)b * S;  // row offset into [B*S, D]
  const int cbase = h * 64;              // head column offset (DK = 64)

  // Q fragments straight from global (each lane loads two 8-element k chunks of its row).
  v16bf qf[2];
#pragma unroll
  for (int i = 0; i < 2; i++) {
    const bf16_t* base = Q + (rowbase + q0 + wid * 16 + l16) * D + cbase + i * 32;
    qf[i] = ld_pair8(base + lh8, base + 16 + lh8);
  }

  const v8f vzero = {0.f, 0.f, 0.f, 0.f, 0.f, 0.f, 0.f, 0.f};
  v8f oacc[4];
#pragma unroll
  for (int j = 0; j < 4; j++) oacc[j] = vzero;
  float mrow[8], srow[8];
#pragma unroll
  for (int r = 0; r < 8; r++) { mrow[r] = -1.0e30f; srow[r] = 0.f; }

  const int key0 = tid >> 1;
  const int dkc  = (tid & 1) * 32;

  for (int kt0 = 0; kt0 < q0 + 64; kt0 += 64) {
    const int d = q0 - kt0;                       // >= 0, multiple of 64
    if (!(d <= 128 || (d & 127) == 0)) continue;  // fully-masked tile -> skip

    {  // K tile [key][dk]
      const bf16_t* src = Kg + (rowbase + kt0 + key0) * D + cbase + dkc;
      bf16_t* dst = &ldsK[key0 * LDK + dkc];
#pragma unroll
      for (int u = 0; u < 4; u++) *(v8bf*)(dst + u * 8) = *(const v8bf*)(src + u * 8);
    }
    {  // V tile transposed [dk][key]
      const bf16_t* src = Vg + (rowbase + kt0 + key0) * D + cbase + dkc;
#pragma unroll
      for (int i = 0; i < 32; i++) ldsVt[(dkc + i) * LDK + key0] = src[i];
    }
    __syncthreads();

    // scores = Q @ K^T  (4 key sub-tiles x 2 dk steps)
    v8f sc[4];
#pragma unroll
    for (int j = 0; j < 4; j++) sc[j] = vzero;
#pragma unroll
    for (int j = 0; j < 4; j++)
#pragma unroll
      for (int ks = 0; ks < 2; ks++) {
        const bf16_t* bp = &ldsK[(j * 16 + l16) * LDK + ks * 32];
        sc[j] = wmma_bf16(qf[ks], ld_pair8(bp + lh8, bp + 16 + lh8), sc[j]);
      }

    // scale + strided/local mask + user mask
#pragma unroll
    for (int j = 0; j < 4; j++) {
      const int jg = kt0 + j * 16 + l16;
#pragma unroll
      for (int r = 0; r < 8; r++) {
        const int ig = q0 + wid * 16 + r + lh8;
        const int dd = ig - jg;
        float v = sc[j][r] * 0.125f;  // 1/sqrt(64)
        const bool ok = (dd >= 0) && ((dd <= 128) || ((dd & 127) == 0));
        if (!ok) v = NEG_SCORE;
        else if (umask[(size_t)ig * S + jg] == 0) v = NEG_SCORE;
        sc[j][r] = v;
      }
    }

    // online softmax: row max (butterfly within 16-lane row group)
    float lm[8];
#pragma unroll
    for (int r = 0; r < 8; r++)
      lm[r] = fmaxf(fmaxf(sc[0][r], sc[1][r]), fmaxf(sc[2][r], sc[3][r]));
#pragma unroll
    for (int off = 1; off <= 8; off <<= 1)
#pragma unroll
      for (int r = 0; r < 8; r++) lm[r] = fmaxf(lm[r], __shfl_xor(lm[r], off, 32));

    float rsum[8];
#pragma unroll
    for (int r = 0; r < 8; r++) {
      const float nm = fmaxf(mrow[r], lm[r]);
      const float corr = __expf(mrow[r] - nm);
      mrow[r] = nm;
      srow[r] *= corr;
      float ps = 0.f;
#pragma unroll
      for (int j = 0; j < 4; j++) {
        const float p = __expf(sc[j][r] - nm);
        sc[j][r] = p;
        ps += p;
      }
      rsum[r] = ps;
#pragma unroll
      for (int j = 0; j < 4; j++) oacc[j][r] *= corr;
    }
#pragma unroll
    for (int off = 1; off <= 8; off <<= 1)
#pragma unroll
      for (int r = 0; r < 8; r++) rsum[r] += __shfl_xor(rsum[r], off, 32);
#pragma unroll
    for (int r = 0; r < 8; r++) srow[r] += rsum[r];

    // C-layout P -> A-layout via per-wave LDS bounce (wave-private, DS in order)
    bf16_t* pbuf = ldsP[wid];
#pragma unroll
    for (int j = 0; j < 4; j++)
#pragma unroll
      for (int r = 0; r < 8; r++)
        pbuf[(r + lh8) * LDK + j * 16 + l16] = (bf16_t)sc[j][r];

    v16bf pf[2];
#pragma unroll
    for (int ks = 0; ks < 2; ks++) {
      const bf16_t* base = &pbuf[l16 * LDK + ks * 32];
      pf[ks] = ld_pair8(base + lh8, base + 16 + lh8);
    }
    // O += P @ V
#pragma unroll
    for (int j = 0; j < 4; j++)
#pragma unroll
      for (int ks = 0; ks < 2; ks++) {
        const bf16_t* bp = &ldsVt[(j * 16 + l16) * LDK + ks * 32];
        oacc[j] = wmma_bf16(pf[ks], ld_pair8(bp + lh8, bp + 16 + lh8), oacc[j]);
      }

    __syncthreads();  // before K/V tiles are overwritten (uniform control flow)
  }

  // normalize and store bf16 output [B*S, D] (head columns)
#pragma unroll
  for (int j = 0; j < 4; j++) {
#pragma unroll
    for (int r = 0; r < 8; r++) {
      const int row = q0 + wid * 16 + r + lh8;
      const int col = cbase + j * 16 + l16;
      O[(rowbase + row) * D + col] = (bf16_t)(oacc[j][r] / srow[r]);
    }
  }
}

// ---------------------------------------------------------------------------
// Host-side launch
// ---------------------------------------------------------------------------
extern "C" void kernel_launch(void* const* d_in, const int* in_sizes, int n_in,
                              void* d_out, int out_size, void* d_ws, size_t ws_size,
                              hipStream_t stream) {
  (void)in_sizes; (void)n_in; (void)out_size; (void)ws_size;

  const int B = 2, S = 2048, D = 1024, H = 16, FF = 4096;
  const int M = B * S;  // 4096

  const float* x    = (const float*)d_in[0];
  const int*   msk  = (const int*)d_in[1];
  const float* Wq   = (const float*)d_in[2];
  const float* bq   = (const float*)d_in[3];
  const float* Wk   = (const float*)d_in[4];
  const float* bk   = (const float*)d_in[5];
  const float* Wv   = (const float*)d_in[6];
  const float* bv   = (const float*)d_in[7];
  const float* Wo   = (const float*)d_in[8];
  const float* bo   = (const float*)d_in[9];
  const float* W1   = (const float*)d_in[10];
  const float* bf1  = (const float*)d_in[11];
  const float* W2   = (const float*)d_in[12];
  const float* bf2  = (const float*)d_in[13];
  const float* g1   = (const float*)d_in[14];
  const float* bl1  = (const float*)d_in[15];
  const float* g2   = (const float*)d_in[16];
  const float* bl2  = (const float*)d_in[17];
  float* out = (float*)d_out;

  // workspace carve (~120 MB total)
  char* w = (char*)d_ws;
  const size_t actB = (size_t)M * D * sizeof(bf16_t);
  bf16_t* ln1 = (bf16_t*)w; w += actB;
  bf16_t* qb  = (bf16_t*)w; w += actB;
  bf16_t* kb  = (bf16_t*)w; w += actB;
  bf16_t* vb  = (bf16_t*)w; w += actB;
  bf16_t* ao  = (bf16_t*)w; w += actB;
  float*  x1  = (float*)w;  w += (size_t)M * D * sizeof(float);
  bf16_t* ln2 = (bf16_t*)w; w += actB;
  bf16_t* ffh = (bf16_t*)w; w += (size_t)M * FF * sizeof(bf16_t);
  const size_t wDD = (size_t)D * D * sizeof(bf16_t);
  bf16_t* Wqt = (bf16_t*)w; w += wDD;
  bf16_t* Wkt = (bf16_t*)w; w += wDD;
  bf16_t* Wvt = (bf16_t*)w; w += wDD;
  bf16_t* Wot = (bf16_t*)w; w += wDD;
  bf16_t* W1t = (bf16_t*)w; w += (size_t)D * FF * sizeof(bf16_t);
  bf16_t* W2t = (bf16_t*)w; w += (size_t)FF * D * sizeof(bf16_t);

  const dim3 blk256(256), blk128(128);
  const dim3 gD(D / GBN, M / GBM);    // (8, 32)
  const dim3 gFF(FF / GBN, M / GBM);  // (32, 32)
  const dim3 gAttn(S / 64, B * H);    // (32, 32)
  const dim3 gTDD(D / 64, D / 64);    // (16, 16)
  const dim3 gT1(FF / 64, D / 64);    // W1 [D,FF] -> [FF,D]
  const dim3 gT2(D / 64, FF / 64);    // W2 [FF,D] -> [D,FF]

  // 0) weight convert + transpose (fp32 [K,N] -> bf16 [N,K])
  hipLaunchKernelGGL(wt_bf16_kernel, gTDD, blk256, 0, stream, Wq, Wqt, D, D);
  hipLaunchKernelGGL(wt_bf16_kernel, gTDD, blk256, 0, stream, Wk, Wkt, D, D);
  hipLaunchKernelGGL(wt_bf16_kernel, gTDD, blk256, 0, stream, Wv, Wvt, D, D);
  hipLaunchKernelGGL(wt_bf16_kernel, gTDD, blk256, 0, stream, Wo, Wot, D, D);
  hipLaunchKernelGGL(wt_bf16_kernel, gT1,  blk256, 0, stream, W1, W1t, D, FF);
  hipLaunchKernelGGL(wt_bf16_kernel, gT2,  blk256, 0, stream, W2, W2t, FF, D);

  // 1) LN1 -> bf16
  hipLaunchKernelGGL(ln_bf16_kernel, dim3(M), blk256, 0, stream, x, g1, bl1, ln1);
  // 2) Q, K, V projections (bf16 out)
  hipLaunchKernelGGL(gemm_bf16_wmma, gD, blk256, 0, stream,
                     ln1, Wqt, bq, (const float*)nullptr, (float*)nullptr, qb,
                     M, D, D, 0);
  hipLaunchKernelGGL(gemm_bf16_wmma, gD, blk256, 0, stream,
                     ln1, Wkt, bk, (const float*)nullptr, (float*)nullptr, kb,
                     M, D, D, 0);
  hipLaunchKernelGGL(gemm_bf16_wmma, gD, blk256, 0, stream,
                     ln1, Wvt, bv, (const float*)nullptr, (float*)nullptr, vb,
                     M, D, D, 0);
  // 3) strided/local flash attention
  hipLaunchKernelGGL(attn_wmma, gAttn, blk128, 0, stream,
                     qb, kb, vb, msk, ao, S, D, H);
  // 4) output projection + residual -> x1 (fp32)
  hipLaunchKernelGGL(gemm_bf16_wmma, gD, blk256, 0, stream,
                     ao, Wot, bo, x, x1, (bf16_t*)nullptr, M, D, D, 0);
  // 5) LN2 -> bf16
  hipLaunchKernelGGL(ln_bf16_kernel, dim3(M), blk256, 0, stream, x1, g2, bl2, ln2);
  // 6) FFN up + exact GELU (bf16 out)
  hipLaunchKernelGGL(gemm_bf16_wmma, gFF, blk256, 0, stream,
                     ln2, W1t, bf1, (const float*)nullptr, (float*)nullptr, ffh,
                     M, FF, D, 1);
  // 7) FFN down + residual -> final fp32 output
  hipLaunchKernelGGL(gemm_bf16_wmma, gD, blk256, 0, stream,
                     ffh, W2t, bf2, x1, out, (bf16_t*)nullptr, M, D, FF, 0);
}
